// MaxVit_GAN_Block_31851477467690
// MI455X (gfx1250) — matrix operand order repaired and checked by reading, not compiled, not run on hardware
//
#include <hip/hip_runtime.h>
#include <hip/hip_bf16.h>
#include <math.h>

// MaxViT block forward for MI455X (gfx1250, wave32, WMMA).
// Heavy projections (FFN, qkv, out-proj, 1x1 convs) -> v_wmma_f32_16x16x32_bf16
// with LDS-staged bf16 tiles; irregular small math stays VALU.

typedef __attribute__((ext_vector_type(16))) __bf16 v16bf;
typedef __attribute__((ext_vector_type(8)))  __bf16 v8bf;
typedef __attribute__((ext_vector_type(8)))  float  v8f;

#define T_TOK 3136   // 56*56 tokens per batch
#define HWV   56
#define CCH   128
#define NB    16
#define KC    128    // K-chunk staged in LDS
#define RS    136    // padded row stride (elements): 128 + 8 (16B pad)

__device__ __forceinline__ float gelu_f(float x) {
  return 0.5f * x * (1.f + erff(x * 0.70710678118654752f));
}
__device__ __forceinline__ float silu_f(float x) {
  return x / (1.f + __expf(-x));
}

// ---------------------------------------------------------------------------
// WMMA GEMM over NCHW activations: Y[b][m][t] = act(sum_k W(m,k) X[b][k][t] + b)
// W(m,k) = W[m*sm + k*sk]  (o-major: sm=Cin,sk=1; c-major: sm=1,sk=Cout)
// Block: 256 threads = 8 waves. Block tile M=64 x N=64, K chunked by 128.
// Wave w: M-tile = w>>1, N-subtiles {2*(w&1), 2*(w&1)+1}. 8 static WMMAs/chunk.
// ---------------------------------------------------------------------------
__global__ __launch_bounds__(256) void gemm_wmma_nchw(
    const float* __restrict__ X, const float* __restrict__ W,
    const float* __restrict__ bias, float* __restrict__ Y,
    int Cin, int Cout, int sm, int sk, int act)
{
  __shared__ __bf16 ldsA[64 * RS];   // weights  [m][k] bf16, padded rows
  __shared__ __bf16 ldsB[64 * RS];   // activ.   [n][k] bf16, padded rows

  const int b    = blockIdx.z;
  const int n0   = blockIdx.x * 64;
  const int m0b  = blockIdx.y * 64;
  const int tid  = threadIdx.x;
  const int wave = tid >> 5;
  const int lane = tid & 31;
  const int half = lane >> 4;
  const int mr   = lane & 15;
  const int mt   = wave >> 1;             // M-tile 0..3
  const int nq   = wave & 1;              // N-half 0..1
  const int m0   = m0b + mt * 16;
  const float* Xb = X + (size_t)b * Cin * T_TOK;
  float* Yb       = Y + (size_t)b * Cout * T_TOK;

  __builtin_prefetch(W, 0, 1);           // global_prefetch_b8

  v8f acc0 = {}, acc1 = {};

  for (int kc = 0; kc < Cin; kc += KC) {
    // ---- stage A: 64 x 128 weights (guard m >= Cout with zeros) ----
    if (sk == 1) {             // o-major: coalesce over k
      for (int idx = tid; idx < 64 * KC; idx += 256) {
        int k = idx & (KC - 1), m = idx >> 7;
        int gm = m0b + m;
        float wv = (gm < Cout) ? W[(size_t)gm * sm + (size_t)(kc + k)] : 0.f;
        ldsA[m * RS + k] = (__bf16)wv;
      }
    } else {                   // c-major: coalesce over m
      for (int idx = tid; idx < 64 * KC; idx += 256) {
        int m = idx & 63, k = idx >> 6;
        int gm = m0b + m;
        float wv = (gm < Cout) ? W[(size_t)(kc + k) * sk + gm] : 0.f;
        ldsA[m * RS + k] = (__bf16)wv;
      }
    }
    // ---- stage B: 128 x 64 activations (coalesced over n) ----
    for (int idx = tid; idx < 64 * KC; idx += 256) {
      int n = idx & 63, k = idx >> 6;
      ldsB[n * RS + k] = (__bf16)Xb[(size_t)(kc + k) * T_TOK + n0 + n];
    }
    __syncthreads();

    if (m0 < Cout) {
      const __bf16* arow  = ldsA + ((mt * 16) + mr) * RS;
      const __bf16* brow0 = ldsB + (nq * 32 + mr) * RS;
      const __bf16* brow1 = ldsB + (nq * 32 + 16 + mr) * RS;
#pragma unroll
      for (int k0 = 0; k0 < KC; k0 += 32) {
        v8bf a_lo = *(const v8bf*)(arow + k0 + half * 16);
        v8bf a_hi = *(const v8bf*)(arow + k0 + half * 16 + 8);
        v16bf afrag = __builtin_shufflevector(a_lo, a_hi,
            0,1,2,3,4,5,6,7,8,9,10,11,12,13,14,15);
        v8bf b0l = *(const v8bf*)(brow0 + k0 + half * 16);
        v8bf b0h = *(const v8bf*)(brow0 + k0 + half * 16 + 8);
        v16bf bf0 = __builtin_shufflevector(b0l, b0h,
            0,1,2,3,4,5,6,7,8,9,10,11,12,13,14,15);
        v8bf b1l = *(const v8bf*)(brow1 + k0 + half * 16);
        v8bf b1h = *(const v8bf*)(brow1 + k0 + half * 16 + 8);
        v16bf bf1 = __builtin_shufflevector(b1l, b1h,
            0,1,2,3,4,5,6,7,8,9,10,11,12,13,14,15);
        acc0 = __builtin_amdgcn_wmma_f32_16x16x32_bf16(
            false, afrag, false, bf0, (short)0, acc0, false, false);
        acc1 = __builtin_amdgcn_wmma_f32_16x16x32_bf16(
            false, afrag, false, bf1, (short)0, acc1, false, false);
      }
    }
    __syncthreads();
  }

  if (m0 >= Cout) return;
  // C/D layout: vgpr i -> M = m0 + i + half*8, N = base + mr
  const int n_a = n0 + nq * 32 + mr;
  const int n_b = n_a + 16;
#pragma unroll
  for (int i = 0; i < 8; ++i) {
    int m = m0 + i + half * 8;
    if (m < Cout) {
      float bv = bias ? bias[m] : 0.f;
      float va = acc0[i] + bv;
      float vb = acc1[i] + bv;
      if (act == 1) { va = gelu_f(va); vb = gelu_f(vb); }
      Yb[(size_t)m * T_TOK + n_a] = va;
      Yb[(size_t)m * T_TOK + n_b] = vb;
    }
  }
}

// ---------------------------------------------------------------------------
// Criss-cross attention: scores + softmax over 112 (column + row stripe)
// ---------------------------------------------------------------------------
__global__ __launch_bounds__(256) void cca_scores(
    const float* __restrict__ Q, const float* __restrict__ K,
    float* __restrict__ A)
{
  int tid = blockIdx.x * blockDim.x + threadIdx.x;
  if (tid >= NB * T_TOK) return;
  int b = tid / T_TOK, t = tid - b * T_TOK;
  int h = t / HWV, w = t - h * HWV;
  const float* Qb = Q + (size_t)b * 16 * T_TOK;
  const float* Kb = K + (size_t)b * 16 * T_TOK;
  float qv[16];
#pragma unroll
  for (int c = 0; c < 16; ++c) qv[c] = Qb[c * T_TOK + t];
  float* Ar = A + ((size_t)b * T_TOK + t) * 112;
  float mx = -1e30f;
  for (int g = 0; g < HWV; ++g) {
    float acc = 0.f; int tt = g * HWV + w;
#pragma unroll
    for (int c = 0; c < 16; ++c) acc += qv[c] * Kb[c * T_TOK + tt];
    if (g == h) acc = -1e30f;
    Ar[g] = acc; mx = fmaxf(mx, acc);
  }
  for (int u = 0; u < HWV; ++u) {
    float acc = 0.f; int tt = h * HWV + u;
#pragma unroll
    for (int c = 0; c < 16; ++c) acc += qv[c] * Kb[c * T_TOK + tt];
    Ar[56 + u] = acc; mx = fmaxf(mx, acc);
  }
  float s = 0.f;
  for (int j = 0; j < 112; ++j) { float e = __expf(Ar[j] - mx); Ar[j] = e; s += e; }
  float inv = 1.f / s;
  for (int j = 0; j < 112; ++j) Ar[j] *= inv;
}

__global__ __launch_bounds__(256) void cca_out(
    const float* __restrict__ V, const float* __restrict__ A,
    const float* __restrict__ Xres, const float* __restrict__ gamma,
    float* __restrict__ Out)
{
  size_t tid = (size_t)blockIdx.x * blockDim.x + threadIdx.x;
  if (tid >= (size_t)NB * CCH * T_TOK) return;
  int t = (int)(tid % T_TOK);
  int c = (int)((tid / T_TOK) % CCH);
  int b = (int)(tid / ((size_t)CCH * T_TOK));
  int h = t / HWV, w = t - h * HWV;
  const float* Vb = V + ((size_t)b * CCH + c) * T_TOK;
  const float* Ar = A + ((size_t)b * T_TOK + t) * 112;
  float acc = 0.f;
  for (int g = 0; g < HWV; ++g) acc += Vb[g * HWV + w] * Ar[g];
  for (int u = 0; u < HWV; ++u) acc += Vb[h * HWV + u] * Ar[56 + u];
  Out[tid] = gamma[0] * acc + Xres[tid];
}

// ---------------------------------------------------------------------------
// BN (inference) + optional act + optional residual. P = [4][C] g,b,m,v.
// ---------------------------------------------------------------------------
__global__ __launch_bounds__(256) void bn_act_res(
    const float* __restrict__ In, const float* __restrict__ Res,
    const float* __restrict__ P, float* __restrict__ Out, int act)
{
  size_t tid = (size_t)blockIdx.x * blockDim.x + threadIdx.x;
  if (tid >= (size_t)NB * CCH * T_TOK) return;
  int c = (int)((tid / T_TOK) % CCH);
  float g = P[c], bb = P[CCH + c], m = P[2 * CCH + c], v = P[3 * CCH + c];
  float y = (In[tid] - m) * (g * rsqrtf(v + 1e-5f)) + bb;
  if (act == 1) y = silu_f(y);
  if (Res) y += Res[tid];
  Out[tid] = y;
}

// ---------------------------------------------------------------------------
// 7x7 window attention (one batch). One block per (window, head).
// ---------------------------------------------------------------------------
__global__ __launch_bounds__(64) void win_attn(
    const float* __restrict__ QKV, const float* __restrict__ bias,
    float* __restrict__ Out, int grid_mode)
{
  const int head = blockIdx.x & 3;
  const int wid  = blockIdx.x >> 2;
  const int nx = wid >> 3, ny = wid & 7;
  __shared__ float q[49][32], k[49][32], v[49][32];

  for (int idx = threadIdx.x; idx < 49 * 32; idx += 64) {
    int i = idx >> 5, d = idx & 31;
    int i1 = i / 7, i2 = i - i1 * 7;
    int h = grid_mode ? (i1 * 8 + nx) : (nx * 7 + i1);
    int w = grid_mode ? (i2 * 8 + ny) : (ny * 7 + i2);
    int t = h * HWV + w;
    int ch = head * 32 + d;
    q[i][d] = QKV[(size_t)(ch) * T_TOK + t] * 0.17677669529663687f;
    k[i][d] = QKV[(size_t)(CCH + ch) * T_TOK + t];
    v[i][d] = QKV[(size_t)(2 * CCH + ch) * T_TOK + t];
  }
  __syncthreads();

  int i = threadIdx.x;
  if (i >= 49) return;
  int i1 = i / 7, i2 = i - i1 * 7;
  float l[49];
  float mx = -1e30f;
  for (int j = 0; j < 49; ++j) {
    float acc = 0.f;
#pragma unroll
    for (int d = 0; d < 32; ++d) acc += q[i][d] * k[j][d];
    int j1 = j / 7, j2 = j - j1 * 7;
    int rel = (i1 - j1 + 6) * 13 + (i2 - j2 + 6);
    acc += bias[rel * 4 + head];
    l[j] = acc; mx = fmaxf(mx, acc);
  }
  float s = 0.f;
  for (int j = 0; j < 49; ++j) { float e = __expf(l[j] - mx); l[j] = e; s += e; }
  float inv = 1.f / s;
  int h = grid_mode ? (i1 * 8 + nx) : (nx * 7 + i1);
  int w = grid_mode ? (i2 * 8 + ny) : (ny * 7 + i2);
  int t = h * HWV + w;
  for (int d = 0; d < 32; ++d) {
    float acc = 0.f;
    for (int j = 0; j < 49; ++j) acc += l[j] * v[j][d];
    Out[(size_t)(head * 32 + d) * T_TOK + t] = acc * inv;
  }
}

// ---------------------------------------------------------------------------
// MBConv pieces
// ---------------------------------------------------------------------------
__global__ __launch_bounds__(256) void dwconv3(
    const float* __restrict__ In, const float* __restrict__ Wd,
    const float* __restrict__ Bd, float* __restrict__ Out)
{
  size_t tid = (size_t)blockIdx.x * blockDim.x + threadIdx.x;
  if (tid >= (size_t)NB * CCH * T_TOK) return;
  int t = (int)(tid % T_TOK);
  int c = (int)((tid / T_TOK) % CCH);
  int b = (int)(tid / ((size_t)CCH * T_TOK));
  int h = t / HWV, w = t - h * HWV;
  const float* I  = In + ((size_t)b * CCH + c) * T_TOK;
  const float* wp = Wd + c * 9;
  float acc = Bd[c];
#pragma unroll
  for (int kh = -1; kh <= 1; ++kh)
#pragma unroll
    for (int kw = -1; kw <= 1; ++kw) {
      int hh = h + kh, ww = w + kw;
      if (hh >= 0 && hh < HWV && ww >= 0 && ww < HWV)
        acc += wp[(kh + 1) * 3 + (kw + 1)] * I[hh * HWV + ww];
    }
  Out[tid] = acc;
}

__global__ __launch_bounds__(256) void mean_hw(
    const float* __restrict__ In, float* __restrict__ S)
{
  int tid = blockIdx.x * blockDim.x + threadIdx.x;
  if (tid >= NB * CCH) return;
  const float* I = In + (size_t)tid * T_TOK;
  float s = 0.f;
  for (int t = 0; t < T_TOK; ++t) s += I[t];
  S[tid] = s * (1.f / (float)T_TOK);
}

__global__ __launch_bounds__(128) void se_gate(
    const float* __restrict__ S, const float* __restrict__ W1,
    const float* __restrict__ W2, float* __restrict__ G)
{
  __shared__ float tbuf[32];
  int b = blockIdx.x;
  int tid = threadIdx.x;
  if (tid < 32) {
    float acc = 0.f;
    for (int c = 0; c < CCH; ++c) acc += S[b * CCH + c] * W1[c * 32 + tid];
    tbuf[tid] = silu_f(acc);
  }
  __syncthreads();
  float acc = 0.f;
#pragma unroll
  for (int j = 0; j < 32; ++j) acc += tbuf[j] * W2[j * CCH + tid];
  G[b * CCH + tid] = 1.f / (1.f + __expf(-acc));
}

__global__ __launch_bounds__(256) void scale_gate(
    float* __restrict__ X, const float* __restrict__ G)
{
  size_t tid = (size_t)blockIdx.x * blockDim.x + threadIdx.x;
  if (tid >= (size_t)NB * CCH * T_TOK) return;
  int c = (int)((tid / T_TOK) % CCH);
  int b = (int)(tid / ((size_t)CCH * T_TOK));
  X[tid] *= G[b * CCH + c];
}

__global__ __launch_bounds__(256) void add_out(
    const float* __restrict__ A, const float* __restrict__ B,
    float* __restrict__ O)
{
  size_t tid = (size_t)blockIdx.x * blockDim.x + threadIdx.x;
  if (tid >= (size_t)NB * CCH * T_TOK) return;
  O[tid] = A[tid] + B[tid];
}

// ---------------------------------------------------------------------------
static inline void launch_gemm(const float* X, const float* W, const float* bias,
                               float* Y, int Cin, int Cout, int nb,
                               int wt_cmajor, int act, hipStream_t s)
{
  dim3 grid(T_TOK / 64, (Cout + 63) / 64, nb);
  int sm = wt_cmajor ? 1 : Cin;
  int sk = wt_cmajor ? Cout : 1;
  gemm_wmma_nchw<<<grid, 256, 0, s>>>(X, W, bias, Y, Cin, Cout, sm, sk, act);
}

extern "C" void kernel_launch(void* const* d_in, const int* in_sizes, int n_in,
                              void* d_out, int out_size, void* d_ws, size_t ws_size,
                              hipStream_t stream)
{
  (void)in_sizes; (void)n_in; (void)out_size; (void)ws_size;
  auto F = [&](int i) { return (const float*)d_in[i]; };
  const float* x0 = F(0);

  // workspace layout (floats). Requires ~119 MB.
  float* ws = (float*)d_ws;
  const size_t ACT = (size_t)NB * CCH * T_TOK;
  float* W0v  = ws;
  float* W1v  = W0v + ACT;
  float* W2v  = W1v + ACT;
  float* Ab   = W2v + ACT;                         // [B][T][112]
  float* Qb   = Ab  + (size_t)NB * T_TOK * 112;    // [B][16][T]
  float* Kb   = Qb  + (size_t)NB * 16 * T_TOK;
  float* H3   = Kb  + (size_t)NB * 16 * T_TOK;     // per-batch [512][T]
  float* QKVb = H3  + (size_t)512 * T_TOK;         // per-batch [384][T]
  float* AOb  = QKVb + (size_t)384 * T_TOK;        // per-batch [128][T]
  float* Sb   = AOb + (size_t)CCH * T_TOK;         // [B][C]
  float* Gb   = Sb + NB * CCH;                     // [B][C]

  const size_t total = ACT;
  const int EB = 256;
  const int egrid = (int)((total + EB - 1) / EB);
  const int sgrid = (NB * T_TOK + EB - 1) / EB;
  const size_t BOF = (size_t)CCH * T_TOK;

  // ---- CCA 1 : x1 = W2v -------------------------------------------------
  launch_gemm(x0, F(1), F(2), Qb, CCH, 16, NB, 0, 0, stream);
  launch_gemm(x0, F(3), F(4), Kb, CCH, 16, NB, 0, 0, stream);
  launch_gemm(x0, F(5), F(6), W1v, CCH, CCH, NB, 0, 0, stream);
  cca_scores<<<sgrid, EB, 0, stream>>>(Qb, Kb, Ab);
  cca_out<<<egrid, EB, 0, stream>>>(W1v, Ab, x0, F(7), W2v);

  // ---- CCA 2 then x2 = bn1(cca2)+x1 -> W0v ------------------------------
  launch_gemm(W2v, F(8),  F(9),  Qb, CCH, 16, NB, 0, 0, stream);
  launch_gemm(W2v, F(10), F(11), Kb, CCH, 16, NB, 0, 0, stream);
  launch_gemm(W2v, F(12), F(13), W1v, CCH, CCH, NB, 0, 0, stream);
  cca_scores<<<sgrid, EB, 0, stream>>>(Qb, Kb, Ab);
  cca_out<<<egrid, EB, 0, stream>>>(W1v, Ab, W2v, F(14), W0v);
  bn_act_res<<<egrid, EB, 0, stream>>>(W0v, W2v, F(15), W0v, 0);

  // ---- FFN1 (per batch): x3=W1v; x4 = bn2(x3)+x3 -> W0v -----------------
  for (int b = 0; b < NB; ++b) {
    launch_gemm(W0v + b * BOF, F(16), F(17), H3, CCH, 512, 1, 1, 1, stream);
    launch_gemm(H3, F(18), F(19), W1v + b * BOF, 512, CCH, 1, 1, 0, stream);
  }
  bn_act_res<<<egrid, EB, 0, stream>>>(W1v, W1v, F(20), W0v, 0);

  // ---- Grid window attention: y1=W1v; y2 = bn3(y1)+y1 -> W0v ------------
  for (int b = 0; b < NB; ++b) {
    launch_gemm(W0v + b * BOF, F(21), nullptr, QKVb, CCH, 384, 1, 1, 0, stream);
    win_attn<<<256, 64, 0, stream>>>(QKVb, F(23), AOb, 1);
    launch_gemm(AOb, F(22), nullptr, W1v + b * BOF, CCH, CCH, 1, 1, 0, stream);
  }
  bn_act_res<<<egrid, EB, 0, stream>>>(W1v, W1v, F(24), W0v, 0);

  // ---- FFN (ffg): y3=W1v; y4 = bn4+res -> W0v ---------------------------
  for (int b = 0; b < NB; ++b) {
    launch_gemm(W0v + b * BOF, F(25), F(26), H3, CCH, 512, 1, 1, 1, stream);
    launch_gemm(H3, F(27), F(28), W1v + b * BOF, 512, CCH, 1, 1, 0, stream);
  }
  bn_act_res<<<egrid, EB, 0, stream>>>(W1v, W1v, F(29), W0v, 0);

  // ---- Block window attention: z1=W1v; z2 = bn5+res -> W0v --------------
  for (int b = 0; b < NB; ++b) {
    launch_gemm(W0v + b * BOF, F(30), nullptr, QKVb, CCH, 384, 1, 1, 0, stream);
    win_attn<<<256, 64, 0, stream>>>(QKVb, F(32), AOb, 0);
    launch_gemm(AOb, F(31), nullptr, W1v + b * BOF, CCH, CCH, 1, 1, 0, stream);
  }
  bn_act_res<<<egrid, EB, 0, stream>>>(W1v, W1v, F(33), W0v, 0);

  // ---- FFN (ffb): z3=W1v; z4 = bn6+res -> W0v ---------------------------
  for (int b = 0; b < NB; ++b) {
    launch_gemm(W0v + b * BOF, F(34), F(35), H3, CCH, 512, 1, 1, 1, stream);
    launch_gemm(H3, F(36), F(37), W1v + b * BOF, 512, CCH, 1, 1, 0, stream);
  }
  bn_act_res<<<egrid, EB, 0, stream>>>(W1v, W1v, F(38), W0v, 0);

  // ---- MBConv -----------------------------------------------------------
  launch_gemm(W0v, F(39), F(40), W1v, CCH, CCH, NB, 0, 0, stream);   // c1
  bn_act_res<<<egrid, EB, 0, stream>>>(W1v, nullptr, F(41), W1v, 1); // bn1+silu
  dwconv3<<<egrid, EB, 0, stream>>>(W1v, F(42), F(43), W2v);
  mean_hw<<<(NB * CCH + EB - 1) / EB, EB, 0, stream>>>(W2v, Sb);
  se_gate<<<NB, 128, 0, stream>>>(Sb, F(44), F(45), Gb);
  scale_gate<<<egrid, EB, 0, stream>>>(W2v, Gb);
  launch_gemm(W2v, F(46), F(47), W1v, CCH, CCH, NB, 0, 0, stream);   // c2
  bn_act_res<<<egrid, EB, 0, stream>>>(W1v, nullptr, F(48), W1v, 0); // bn2
  add_out<<<egrid, EB, 0, stream>>>(W1v, W0v, (float*)d_out);
}
